// MultiHeadAttention_55585466745451
// MI455X (gfx1250) — matrix operand order repaired and checked by reading, not compiled, not run on hardware
//
#include <hip/hip_runtime.h>

// Problem constants (from reference)
#define BATCH  2
#define SEQ    2048
#define DMODEL 2048
#define NHEAD  16
#define HDIM   128
#define MTOT   (BATCH * SEQ)   // 4096 rows in [B*S, D] view

typedef __attribute__((ext_vector_type(16))) __bf16 v16bf;
typedef __attribute__((ext_vector_type(8)))  __bf16 v8bf;
typedef __attribute__((ext_vector_type(8)))  float  v8f;
typedef unsigned int u32x4 __attribute__((ext_vector_type(4)));
typedef int          i32x8 __attribute__((ext_vector_type(8)));
typedef int          i32x4 __attribute__((ext_vector_type(4)));

union BF16x16 { v16bf v; v8bf h[2]; };

__device__ __forceinline__ unsigned short f2bf(float x) {
  unsigned u = __float_as_uint(x);
  unsigned r = u + 0x7FFFu + ((u >> 16) & 1u);   // round-to-nearest-even
  return (unsigned short)(r >> 16);
}

// Low 32 bits of a generic LDS pointer == LDS byte offset (aperture rule).
__device__ __forceinline__ unsigned lds_of(const void* p) {
  return (unsigned)(unsigned long long)p;
}

// ---------------------------------------------------------------- TDM helper
// Post a 2D tensor tile load (bf16 elements) Global -> LDS via the Tensor
// Data Mover.  D# layout per CDNA5 ISA 8.3/8.4:
//   group0: [1:0]=count(1), [63:32]=lds_addr, [120:64]=global_addr, [127:126]=type(2)
//   group1: [17:16]=data_size(1 => 2 bytes), [79:48]=tensor_dim0,
//           [111:80]=tensor_dim1, [127:112]=tile_dim0, [143:128]=tile_dim1,
//           [207:160]=tensor_dim0_stride
// groups 2/3 zero (<=2D tensor).  Completion tracked with TENSORcnt.
__device__ __forceinline__ void tdm_load_2d(unsigned lds_addr, const void* gaddr,
                                            unsigned tile_d0, unsigned tile_d1,
                                            unsigned tensor_d0, unsigned tensor_d1,
                                            unsigned stride_d0) {
  unsigned long long ga = (unsigned long long)gaddr;
  u32x4 g0;
  g0[0] = 1u;                                                  // count=1, user D#
  g0[1] = lds_addr;                                            // LDS byte address
  g0[2] = (unsigned)(ga & 0xFFFFFFFFu);                        // global_addr[31:0]
  g0[3] = (unsigned)((ga >> 32) & 0x01FFFFFFu) | (2u << 30);   // addr[56:32] | type=2
  i32x8 g1;
  g1[0] = (int)(1u << 16);                                     // data_size = 2 bytes
  g1[1] = (int)((tensor_d0 & 0xFFFFu) << 16);                  // tensor_dim0[15:0]
  g1[2] = (int)(((tensor_d0 >> 16) & 0xFFFFu) |
                ((tensor_d1 & 0xFFFFu) << 16));                // dim0[31:16] | dim1[15:0]
  g1[3] = (int)(((tensor_d1 >> 16) & 0xFFFFu) |
                ((tile_d0 & 0xFFFFu) << 16));                  // dim1[31:16] | tile_dim0
  g1[4] = (int)(tile_d1 & 0xFFFFu);                            // tile_dim1 (tile_dim2=0)
  g1[5] = (int)stride_d0;                                      // dim0_stride[31:0]
  g1[6] = 0;                                                   // dim0_stride[47:32], dim1_stride[15:0]
  g1[7] = 0;
  i32x4 gz4 = {0, 0, 0, 0};
  i32x8 gz8 = {0, 0, 0, 0, 0, 0, 0, 0};
  __builtin_amdgcn_tensor_load_to_lds(g0, g1, gz4, gz4, gz8, 0);
}

// ---------------------------------------------------------------- cast kernel
__global__ void __launch_bounds__(256)
cast_f32_to_bf16(const float* __restrict__ in, unsigned short* __restrict__ out, int n) {
  int i = (blockIdx.x * blockDim.x + threadIdx.x) * 4;
  if (i < n) {
    float4 f = *(const float4*)(in + i);
    unsigned long long p =
        (unsigned long long)f2bf(f.x) |
        ((unsigned long long)f2bf(f.y) << 16) |
        ((unsigned long long)f2bf(f.z) << 32) |
        ((unsigned long long)f2bf(f.w) << 48);
    *(unsigned long long*)(out + i) = p;
  }
}

// ---------------------------------------------------------------- GEMM kernel
// C[m,n] = sum_k A[m,k] * W[n,k] + bias[n]      (i.e. X @ W^T + b)
// 256 threads = 8 waves, 128x128 tile, K in 32-steps, TDM double-buffered LDS.
// MODE 0: bf16 head-major [B,H,S,HD] output;  MODE 1: fp32 flat [M,N] output.
template <int MODE>
__global__ void __launch_bounds__(256)
gemm_bf16(const unsigned short* __restrict__ A,
          const unsigned short* __restrict__ W,
          const float* __restrict__ bias,
          void* __restrict__ dst) {
  __shared__ __align__(16) unsigned short lA[2][128][32];
  __shared__ __align__(16) unsigned short lB[2][128][32];

  const int tid  = threadIdx.x;
  const int lane = tid & 31;
  const int wave = tid >> 5;
  const int hi   = (lane >> 4) & 1;
  const int l16  = lane & 15;
  const int m0   = blockIdx.y * 128;
  const int n0   = blockIdx.x * 128;
  const int wm   = (wave & 3) * 32;
  const int wn   = (wave >> 2) * 64;
  const int NK   = DMODEL / 32;   // 64 K-steps

  // prologue: post first pair of tiles
  if (wave == 0) {
    tdm_load_2d(lds_of(&lA[0][0][0]), A + (size_t)m0 * DMODEL,
                32, 128, DMODEL, MTOT, DMODEL);
    tdm_load_2d(lds_of(&lB[0][0][0]), W + (size_t)n0 * DMODEL,
                32, 128, DMODEL, DMODEL, DMODEL);
  }

  const v8f vzero = {0.f, 0.f, 0.f, 0.f, 0.f, 0.f, 0.f, 0.f};
  v8f acc[2][4];
#pragma unroll
  for (int i = 0; i < 2; i++)
#pragma unroll
    for (int j = 0; j < 4; j++) acc[i][j] = vzero;

  for (int it = 0; it < NK; it++) {
    if (wave == 0) __builtin_amdgcn_s_wait_tensorcnt(0);  // current tiles landed
    __syncthreads();                                      // visible to all waves
    if (wave == 0 && it + 1 < NK) {
      // post next tiles into the other buffer (its readers passed the barrier)
      const int kk = (it + 1) * 32;
      tdm_load_2d(lds_of(&lA[(it + 1) & 1][0][0]),
                  A + (size_t)m0 * DMODEL + kk, 32, 128, DMODEL, MTOT, DMODEL);
      tdm_load_2d(lds_of(&lB[(it + 1) & 1][0][0]),
                  W + (size_t)n0 * DMODEL + kk, 32, 128, DMODEL, DMODEL, DMODEL);
    }
    const unsigned short (*cA)[32] = lA[it & 1];
    const unsigned short (*cB)[32] = lB[it & 1];

    BF16x16 afrag[2], bfrag[4];
#pragma unroll
    for (int mi = 0; mi < 2; mi++) {
      // A-frag 16x32: lane-lo K 0..7,16..23 ; lane-hi K 8..15,24..31
      const unsigned short* p = &cA[wm + mi * 16 + l16][hi * 8];
      afrag[mi].h[0] = *(const v8bf*)p;
      afrag[mi].h[1] = *(const v8bf*)(p + 16);
    }
#pragma unroll
    for (int ni = 0; ni < 4; ni++) {
      // B-frag 32x16: lane-lo K 0..7,16..23 ; lane-hi K 8..15,24..31
      const unsigned short* p = &cB[wn + ni * 16 + l16][hi * 8];
      bfrag[ni].h[0] = *(const v8bf*)p;
      bfrag[ni].h[1] = *(const v8bf*)(p + 16);
    }
#pragma unroll
    for (int mi = 0; mi < 2; mi++)
#pragma unroll
      for (int ni = 0; ni < 4; ni++)
        acc[mi][ni] = __builtin_amdgcn_wmma_f32_16x16x32_bf16(
            false, afrag[mi].v, false, bfrag[ni].v, (short)0, acc[mi][ni],
            false, false);
  }

  // epilogue: C/D layout — VGPR r holds (M = r + 8*hi, N = l16)
#pragma unroll
  for (int mi = 0; mi < 2; mi++) {
#pragma unroll
    for (int ni = 0; ni < 4; ni++) {
      const int n  = n0 + wn + ni * 16 + l16;
      const float bv = bias[n];
#pragma unroll
      for (int r = 0; r < 8; r++) {
        const int m  = m0 + wm + mi * 16 + r + hi * 8;
        const float val = acc[mi][ni][r] + bv;
        if (MODE == 0) {
          const int b = m >> 11, s = m & 2047, h = n >> 7, d = n & 127;
          const size_t idx = (((size_t)(b * NHEAD + h) * SEQ) + s) * HDIM + d;
          ((unsigned short*)dst)[idx] = f2bf(val);
        } else {
          ((float*)dst)[(size_t)m * DMODEL + n] = val;
        }
      }
    }
  }
}

// ------------------------------------------------------- flash-attention kernel
// Grid: (SEQ/128, B*H). Block: 256 threads = 8 waves; each wave owns 16 query
// rows with Q resident in registers.  32-key steps: K tile via TDM (double
// buffered), V tile software-pipelined through registers into a transposed LDS
// buffer, QK^T and P@V on WMMA, online softmax in fp32.
__global__ void __launch_bounds__(256)
flash_attn(const unsigned short* __restrict__ Q,   // [B*H, S, HD] bf16
           const unsigned short* __restrict__ K,
           const unsigned short* __restrict__ V,
           const float* __restrict__ mask,         // [B, 1, S, S] fp32
           unsigned short* __restrict__ O) {       // [B*S, D] bf16
  __shared__ __align__(16) unsigned short lK[2][32][128];   // key tiles (TDM)
  __shared__ __align__(16) unsigned short lVt[2][128][32];  // value tiles, transposed
  __shared__ __align__(16) unsigned short lP[8][16][32];    // per-wave P scratch

  const int tid  = threadIdx.x;
  const int lane = tid & 31;
  const int wave = tid >> 5;
  const int hi   = (lane >> 4) & 1;
  const int l16  = lane & 15;
  const int bh   = blockIdx.y;
  const int b    = bh >> 4;
  const int h    = bh & 15;
  const int q0   = blockIdx.x * 128 + wave * 16;

  const size_t headoff = (size_t)bh * SEQ * HDIM;
  const unsigned short* Qp = Q + headoff;
  const unsigned short* Kp = K + headoff;
  const unsigned short* Vp = V + headoff;

  const int krow = tid >> 3;        // 0..31  key row within tile
  const int kseg = (tid & 7) * 16;  // d segment
  const int NT   = SEQ / 32;        // 64 key steps

  // prologue: post K tile 0 via TDM, pull V tile 0 into registers, scatter it
  if (wave == 0)
    tdm_load_2d(lds_of(&lK[0][0][0]), Kp, HDIM, 32, HDIM, SEQ, HDIM);
  uint4 vr0, vr1;
  {
    const unsigned short* vp = Vp + (size_t)krow * HDIM + kseg;
    vr0 = *(const uint4*)vp;
    vr1 = *(const uint4*)(vp + 8);
    const unsigned short* s0 = (const unsigned short*)&vr0;
    const unsigned short* s1 = (const unsigned short*)&vr1;
#pragma unroll
    for (int e = 0; e < 8; e++) lVt[0][kseg + e][krow] = s0[e];
#pragma unroll
    for (int e = 0; e < 8; e++) lVt[0][kseg + 8 + e][krow] = s1[e];
  }

  // Q fragments: rows q0..q0+15, HD split into 4 chunks of K=32
  BF16x16 qf[4];
  {
    const unsigned short* qrow = Qp + (size_t)(q0 + l16) * HDIM;
#pragma unroll
    for (int c = 0; c < 4; c++) {
      const unsigned short* p = qrow + c * 32 + hi * 8;
      qf[c].h[0] = *(const v8bf*)p;
      qf[c].h[1] = *(const v8bf*)(p + 16);
    }
  }

  const v8f vzero = {0.f, 0.f, 0.f, 0.f, 0.f, 0.f, 0.f, 0.f};
  v8f oacc[8];
#pragma unroll
  for (int t = 0; t < 8; t++) oacc[t] = vzero;
  float mrow[8], lrow[8];
#pragma unroll
  for (int r = 0; r < 8; r++) { mrow[r] = -1e30f; lrow[r] = 0.f; }

  const float scale = 0.08838834764831845f;  // 1/sqrt(HD)

  for (int it = 0; it < NT; it++) {
    const int kb  = it * 32;
    const int buf = it & 1;

    if (wave == 0) __builtin_amdgcn_s_wait_tensorcnt(0);  // K tile `it` landed
    __syncthreads();  // K visible; V stores of tile `it` (prev iter) visible

    if (it + 1 < NT) {
      if (wave == 0)
        tdm_load_2d(lds_of(&lK[(it + 1) & 1][0][0]),
                    Kp + (size_t)(it + 1) * 32 * HDIM, HDIM, 32, HDIM, SEQ, HDIM);
      const unsigned short* vp = Vp + (size_t)((it + 1) * 32 + krow) * HDIM + kseg;
      vr0 = *(const uint4*)vp;             // issue early, lands during compute
      vr1 = *(const uint4*)(vp + 8);
      if (it + 2 < NT)
        __builtin_prefetch(Vp + (size_t)((it + 2) * 32 + krow) * HDIM + kseg, 0, 1);
    }

    // ---- scores: S(16x32) = Q(16x128) @ K^T, two N-tiles of 16 keys
    v8f s0 = vzero, s1 = vzero;
#pragma unroll
    for (int c = 0; c < 4; c++) {
      BF16x16 bk0, bk1;
      const unsigned short* p0 = &lK[buf][l16][c * 32 + hi * 8];
      bk0.h[0] = *(const v8bf*)p0;
      bk0.h[1] = *(const v8bf*)(p0 + 16);
      const unsigned short* p1 = &lK[buf][16 + l16][c * 32 + hi * 8];
      bk1.h[0] = *(const v8bf*)p1;
      bk1.h[1] = *(const v8bf*)(p1 + 16);
      s0 = __builtin_amdgcn_wmma_f32_16x16x32_bf16(false, qf[c].v, false, bk0.v,
                                                   (short)0, s0, false, false);
      s1 = __builtin_amdgcn_wmma_f32_16x16x32_bf16(false, qf[c].v, false, bk1.v,
                                                   (short)0, s1, false, false);
    }

    // ---- online softmax (C/D striping: row = r + 8*hi, col = l16 / 16+l16)
#pragma unroll
    for (int r = 0; r < 8; r++) {
      const int qrow = q0 + r + hi * 8;
      const float* mp = mask + ((size_t)b * SEQ + qrow) * SEQ + kb + l16;
      float v0 = s0[r] * scale + mp[0];
      float v1 = s1[r] * scale + mp[16];
      float mx = fmaxf(v0, v1);
#pragma unroll
      for (int off = 8; off >= 1; off >>= 1)
        mx = fmaxf(mx, __shfl_xor(mx, off, 16));
      const float newm  = fmaxf(mrow[r], mx);
      const float alpha = __expf(mrow[r] - newm);
      const float p0 = __expf(v0 - newm);
      const float p1 = __expf(v1 - newm);
      float sum = p0 + p1;
#pragma unroll
      for (int off = 8; off >= 1; off >>= 1)
        sum += __shfl_xor(sum, off, 16);
      lrow[r] = lrow[r] * alpha + sum;
      mrow[r] = newm;
#pragma unroll
      for (int t = 0; t < 8; t++) oacc[t][r] *= alpha;
      const int prow = r + hi * 8;
      lP[wave][prow][l16]      = f2bf(p0);
      lP[wave][prow][16 + l16] = f2bf(p1);
    }

    // ---- P fragment (A-layout 16x32) from per-wave scratch
    BF16x16 pf;
    {
      const unsigned short* p = &lP[wave][l16][hi * 8];
      pf.h[0] = *(const v8bf*)p;
      pf.h[1] = *(const v8bf*)(p + 16);
    }

    // ---- O(16x128) += P(16x32) @ V(32x128), 8 N-tiles of 16 dims
#pragma unroll
    for (int t = 0; t < 8; t++) {
      BF16x16 bv;
      const unsigned short* p = &lVt[buf][t * 16 + l16][hi * 8];
      bv.h[0] = *(const v8bf*)p;
      bv.h[1] = *(const v8bf*)(p + 16);
      oacc[t] = __builtin_amdgcn_wmma_f32_16x16x32_bf16(
          false, pf.v, false, bv.v, (short)0, oacc[t], false, false);
    }

    // ---- scatter V tile it+1 (loaded above) into the other buffer
    if (it + 1 < NT) {
      const unsigned short* s0p = (const unsigned short*)&vr0;
      const unsigned short* s1p = (const unsigned short*)&vr1;
      const int nbuf = (it + 1) & 1;
#pragma unroll
      for (int e = 0; e < 8; e++) lVt[nbuf][kseg + e][krow] = s0p[e];
#pragma unroll
      for (int e = 0; e < 8; e++) lVt[nbuf][kseg + 8 + e][krow] = s1p[e];
    }
  }

  // ---- normalize and write context: O[b*S+qrow][h*HD + t*16 + l16]
#pragma unroll
  for (int t = 0; t < 8; t++) {
#pragma unroll
    for (int r = 0; r < 8; r++) {
      const int qrow = q0 + r + hi * 8;
      const float val = oacc[t][r] / lrow[r];
      const size_t idx =
          (size_t)(b * SEQ + qrow) * DMODEL + h * HDIM + t * 16 + l16;
      O[idx] = f2bf(val);
    }
  }
}

// ---------------------------------------------------------------- launcher
extern "C" void kernel_launch(void* const* d_in, const int* in_sizes, int n_in,
                              void* d_out, int out_size, void* d_ws, size_t ws_size,
                              hipStream_t stream) {
  const float* hs   = (const float*)d_in[0];
  const float* mask = (const float*)d_in[1];
  const float* Wq   = (const float*)d_in[2];
  const float* bq   = (const float*)d_in[3];
  const float* Wk   = (const float*)d_in[4];
  const float* bk   = (const float*)d_in[5];
  const float* Wv   = (const float*)d_in[6];
  const float* bv   = (const float*)d_in[7];
  const float* Wo   = (const float*)d_in[8];
  const float* bo   = (const float*)d_in[9];

  const size_t XE = (size_t)MTOT * DMODEL;    // 8,388,608 elems
  const size_t WE = (size_t)DMODEL * DMODEL;  // 4,194,304 elems

  char* ws = (char*)d_ws;
  unsigned short* Xb  = (unsigned short*)ws; ws += XE * 2;   // hidden bf16
  unsigned short* Wqb = (unsigned short*)ws; ws += WE * 2;
  unsigned short* Wkb = (unsigned short*)ws; ws += WE * 2;
  unsigned short* Wvb = (unsigned short*)ws; ws += WE * 2;
  unsigned short* Wob = (unsigned short*)ws; ws += WE * 2;
  unsigned short* Qh  = (unsigned short*)ws; ws += XE * 2;   // [B,H,S,HD]
  unsigned short* Kh  = (unsigned short*)ws; ws += XE * 2;
  unsigned short* Vh  = (unsigned short*)ws; ws += XE * 2;
  unsigned short* Ob  = (unsigned short*)ws; ws += XE * 2;   // context [B*S, D]

  // 1) fp32 -> bf16 casts
  cast_f32_to_bf16<<<(int)(XE / 4 / 256), 256, 0, stream>>>(hs, Xb, (int)XE);
  cast_f32_to_bf16<<<(int)(WE / 4 / 256), 256, 0, stream>>>(Wq, Wqb, (int)WE);
  cast_f32_to_bf16<<<(int)(WE / 4 / 256), 256, 0, stream>>>(Wk, Wkb, (int)WE);
  cast_f32_to_bf16<<<(int)(WE / 4 / 256), 256, 0, stream>>>(Wv, Wvb, (int)WE);
  cast_f32_to_bf16<<<(int)(WE / 4 / 256), 256, 0, stream>>>(Wo, Wob, (int)WE);

  // 2) QKV projections (head-major bf16 outputs)
  dim3 ggrid(DMODEL / 128, MTOT / 128);
  gemm_bf16<0><<<ggrid, 256, 0, stream>>>(Xb, Wqb, bq, Qh);
  gemm_bf16<0><<<ggrid, 256, 0, stream>>>(Xb, Wkb, bk, Kh);
  gemm_bf16<0><<<ggrid, 256, 0, stream>>>(Xb, Wvb, bv, Vh);

  // 3) fused attention
  flash_attn<<<dim3(SEQ / 128, BATCH * NHEAD), 256, 0, stream>>>(Qh, Kh, Vh,
                                                                 mask, Ob);

  // 4) output projection, fp32 result
  gemm_bf16<1><<<ggrid, 256, 0, stream>>>(Ob, Wob, bo, (float*)d_out);
}
